// MultiHeadAttention_86655260164157
// MI455X (gfx1250) — compile-verified, matrix operand order checked by reading, not run on hardware
//
#include <hip/hip_runtime.h>
#include <hip/hip_bf16.h>
#include <math.h>

// MI455X / gfx1250: wave32, WMMA 16x16x32 bf16 with fp32 accumulate.
typedef __attribute__((ext_vector_type(16))) __bf16 v16bf;
typedef __attribute__((ext_vector_type(8)))  float  v8f;

constexpr int SEQ   = 2048;
constexpr int DIM   = 1024;
constexpr int NHEAD = 16;
constexpr int HDIM  = 32;   // rope head dim; exactly one WMMA K step

__device__ __forceinline__ unsigned short f32_to_bf16(float f) {
  unsigned int u = __float_as_uint(f);
  unsigned int r = (u + 0x7FFFu + ((u >> 16) & 1u)) >> 16;  // RNE
  return (unsigned short)r;
}

// 16-bit A-matrix 16x32 fragment K-map (also used for B 32x16, k index):
// lane<16 : e=0..7 -> K=e        ; e=8..15 -> K=16+(e-8)
// lane>=16: e=0..7 -> K=8+e      ; e=8..15 -> K=24+(e-8)
__device__ __forceinline__ int kmap(int e, int lane) {
  return ((e < 8) ? e : (e + 8)) + ((lane & 16) ? 8 : 0);
}

__device__ __forceinline__ v8f wmma_bf16(v16bf a, v16bf b, v8f c) {
  return __builtin_amdgcn_wmma_f32_16x16x32_bf16(false, a, false, b,
                                                 (short)0, c, false, false);
}

// ---------------- pack fp32 row-major (R x C) -> bf16 A-fragments ----------
// dst[tileR][tileK][lane][e], tileR=R/16, tileK=C/32. One thread per (tile,lane).
__global__ void pack_a_kernel(const float* __restrict__ src,
                              unsigned short* __restrict__ dst, int C) {
  int idx  = blockIdx.x * blockDim.x + threadIdx.x;
  int lane = idx & 31;
  int t    = idx >> 5;
  int ktiles = C >> 5;
  int tr = t / ktiles, tk = t % ktiles;
  int row = tr * 16 + (lane & 15);
  unsigned short* out = dst + (((size_t)t * 32 + lane) * 16);
  const float* srow = src + (size_t)row * C + tk * 32;
#pragma unroll
  for (int e = 0; e < 16; ++e) out[e] = f32_to_bf16(srow[kmap(e, lane)]);
}

// ------------- pack weight W[o,d] (DIM x DIM) -> bf16 B-fragments ----------
// B[k=d, n=o] = W[n,k]; dst[tileN][tileK][lane][e], tileN=64, tileK=32.
__global__ void pack_w_kernel(const float* __restrict__ W,
                              unsigned short* __restrict__ dst) {
  int idx  = blockIdx.x * blockDim.x + threadIdx.x;
  int lane = idx & 31;
  int t    = idx >> 5;             // tn*32 + tk
  int tk = t & 31, tn = t >> 5;
  int n = tn * 16 + (lane & 15);
  unsigned short* out = dst + (((size_t)t * 32 + lane) * 16);
  const float* wrow = W + (size_t)n * DIM + tk * 32;
#pragma unroll
  for (int e = 0; e < 16; ++e) out[e] = f32_to_bf16(wrow[kmap(e, lane)]);
}

// ---------- GEMM: C(fp32) = Afrag x Bfrag, one wave per 32(M)x64(N) --------
// Per K-step: 2 A frags + 4 B frags -> 8 WMMAs (11 FLOP/byte vs 4 unblocked).
// Grid: (Mtiles/2)*(Ntiles/4) = 64*16 = 1024 waves = 128 blocks x 8 waves.
__global__ __launch_bounds__(256)
void gemm_bf16_kernel(const unsigned short* __restrict__ A,
                      const unsigned short* __restrict__ B,
                      float* __restrict__ C,
                      int Ktiles, int N) {
  int lane = threadIdx.x & 31;
  int wib  = __builtin_amdgcn_readfirstlane((int)(threadIdx.x >> 5));
  int wid  = blockIdx.x * 8 + wib;        // wave-uniform
  int tn4 = wid & 15;                     // 16 supertiles of 64 cols
  int tm2 = wid >> 4;                     // 64 supertiles of 32 rows
  size_t tstride = (size_t)Ktiles * 512;  // ushorts per 16-row/col tile
  const unsigned short* Ab0 = A + (size_t)(tm2 * 2) * tstride + lane * 16;
  const unsigned short* Ab1 = Ab0 + tstride;
  const unsigned short* Bb  = B + (size_t)(tn4 * 4) * tstride + lane * 16;
  v8f z = {0.f,0.f,0.f,0.f,0.f,0.f,0.f,0.f};
  v8f acc[8];
#pragma unroll
  for (int i = 0; i < 8; ++i) acc[i] = z;
  for (int kk = 0; kk < Ktiles; ++kk) {
    size_t o = (size_t)kk * 512;
    v16bf a0 = *(const v16bf*)(Ab0 + o);
    v16bf a1 = *(const v16bf*)(Ab1 + o);
#pragma unroll
    for (int nb = 0; nb < 4; ++nb) {
      v16bf b = *(const v16bf*)(Bb + (size_t)nb * tstride + o);
      acc[nb]     = wmma_bf16(a0, b, acc[nb]);
      acc[4 + nb] = wmma_bf16(a1, b, acc[4 + nb]);
    }
  }
  int ncol  = tn4 * 64 + (lane & 15);
  int mbase = tm2 * 32 + ((lane >> 4) << 3);   // +8 for upper half
#pragma unroll
  for (int nb = 0; nb < 4; ++nb) {
#pragma unroll
    for (int r = 0; r < 8; ++r) {
      C[(size_t)(mbase + r) * N + ncol + nb * 16]      = acc[nb][r];
      C[(size_t)(mbase + 16 + r) * N + ncol + nb * 16] = acc[4 + nb][r];
    }
  }
}

// -------- RoPE(interleaved) on Q,K then pack: Q->A-frags, K->B-frags -------
// Per (rope-head h2 in 0..31, 16-row tile, lane). A(16xM,32xK) and B(32xK,16xN)
// have identical (row,d) <-> (lane,e) mapping here, so one kernel does both.
__global__ void rope_pack_qk_kernel(const float* __restrict__ Qf,
                                    const float* __restrict__ Kf,
                                    unsigned short* __restrict__ AqF,
                                    unsigned short* __restrict__ BkF) {
  int idx  = blockIdx.x * blockDim.x + threadIdx.x;  // 32*128*32
  int lane = idx & 31;
  int t    = idx >> 5;            // h2*128 + tile
  int h2   = t >> 7;
  int tile = t & 127;
  int row  = tile * 16 + (lane & 15);
  int kbase = (lane & 16) ? 8 : 0;
  unsigned short* aout = AqF + (((size_t)t * 32 + lane) * 16);
  unsigned short* bout = BkF + (((size_t)t * 32 + lane) * 16);
  const float* qrow = Qf + (size_t)row * DIM + h2 * HDIM;
  const float* krow = Kf + (size_t)row * DIM + h2 * HDIM;
#pragma unroll
  for (int e = 0; e < 16; e += 2) {
    int d0 = ((e < 8) ? e : e + 8) + kbase;      // even; d1 = d0+1
    int p  = d0 >> 1;                            // freq index 0..15
    float inv = __powf(10000.0f, -(float)p * (1.0f / 16.0f));
    float ang = (float)row * inv;
    float sn, cs;
    __sincosf(ang, &sn, &cs);
    float x1 = qrow[d0], x2 = qrow[d0 + 1];
    aout[e]     = f32_to_bf16(x1 * cs - x2 * sn);
    aout[e + 1] = f32_to_bf16(x1 * sn + x2 * cs);
    float y1 = krow[d0], y2 = krow[d0 + 1];
    bout[e]     = f32_to_bf16(y1 * cs - y2 * sn);
    bout[e + 1] = f32_to_bf16(y1 * sn + y2 * cs);
  }
}

// --------------- pack V -> B-fragments for P@V: B[k=j, n=vdim] -------------
// dst[h][j32][dt][lane][e]; h=16, j32=64 (32-key steps), dt=4 (64 v dims).
__global__ void pack_v_kernel(const float* __restrict__ Vf,
                              unsigned short* __restrict__ BvF) {
  int idx  = blockIdx.x * blockDim.x + threadIdx.x;  // 16*64*4*32
  int lane = idx & 31;
  int t    = idx >> 5;             // h*256 + j32*4 + dt
  int dt  = t & 3;
  int j32 = (t >> 2) & 63;
  int h   = t >> 8;
  int col = h * 64 + dt * 16 + (lane & 15);
  int jb  = j32 * 32 + ((lane & 16) ? 8 : 0);
  unsigned short* out = BvF + (((size_t)t * 32 + lane) * 16);
#pragma unroll
  for (int e = 0; e < 16; ++e) {
    int j = jb + ((e < 8) ? e : e + 8);
    out[e] = f32_to_bf16(Vf[(size_t)j * DIM + col]);
  }
}

// ------------------------- flash attention per wave ------------------------
// One wave handles (h2, i_tile): 16 queries x full causal key range.
__global__ __launch_bounds__(256)
void flash_kernel(const unsigned short* __restrict__ AqF,
                  const unsigned short* __restrict__ BkF,
                  const unsigned short* __restrict__ BvF,
                  float* __restrict__ attn) {
  __shared__ float lds[8][512];          // 2KB P-staging per wave
  int lane = threadIdx.x & 31;
  int wib  = __builtin_amdgcn_readfirstlane((int)(threadIdx.x >> 5));
  int gw   = blockIdx.x * 8 + wib;       // wave-uniform
  int i_tile = gw & 127;
  int h2     = gw >> 7;
  int h = h2 >> 1, c = h2 & 1;
  int half = lane >> 4;
  float* plds = &lds[wib][0];

  const v16bf aq = *(const v16bf*)(AqF + (((size_t)gw * 32 + lane) * 16));
  v8f o0 = {0.f,0.f,0.f,0.f,0.f,0.f,0.f,0.f};
  v8f o1 = o0, o2 = o0, o3 = o0;
  float mrow[8], lrow[8];
#pragma unroll
  for (int r = 0; r < 8; ++r) { mrow[r] = -1e30f; lrow[r] = 0.f; }

  const float scale = 0.17677669529663687f;   // 1/sqrt(32)
  int nsteps = (i_tile >> 1) + 1;             // 32-key steps, causal bound

  for (int j32 = 0; j32 < nsteps; ++j32) {
    const v16bf bk0 = *(const v16bf*)(BkF + ((((size_t)h2 * 128 + 2*j32    ) * 32 + lane) * 16));
    const v16bf bk1 = *(const v16bf*)(BkF + ((((size_t)h2 * 128 + 2*j32 + 1) * 32 + lane) * 16));
    v8f z = {0.f,0.f,0.f,0.f,0.f,0.f,0.f,0.f};
    v8f s0 = wmma_bf16(aq, bk0, z);
    v8f s1 = wmma_bf16(aq, bk1, z);
    int j0 = j32 * 32 + (lane & 15);
#pragma unroll
    for (int r = 0; r < 8; ++r) {
      int i = i_tile * 16 + r + 8 * half;
      float a = (j0      <= i) ? s0[r] * scale : -1e30f;
      float b = (j0 + 16 <= i) ? s1[r] * scale : -1e30f;
      float mx = fmaxf(a, b);
      mx = fmaxf(mx, __shfl_xor(mx, 1, 32));
      mx = fmaxf(mx, __shfl_xor(mx, 2, 32));
      mx = fmaxf(mx, __shfl_xor(mx, 4, 32));
      mx = fmaxf(mx, __shfl_xor(mx, 8, 32));
      float mn   = fmaxf(mrow[r], mx);
      float corr = __expf(mrow[r] - mn);
      float p0 = __expf(a - mn), p1 = __expf(b - mn);
      float ps = p0 + p1;
      ps += __shfl_xor(ps, 1, 32);
      ps += __shfl_xor(ps, 2, 32);
      ps += __shfl_xor(ps, 4, 32);
      ps += __shfl_xor(ps, 8, 32);
      lrow[r] = lrow[r] * corr + ps;
      mrow[r] = mn;
      o0[r] *= corr; o1[r] *= corr; o2[r] *= corr; o3[r] *= corr;
      plds[(r + 8 * half) * 32 + (lane & 15)]      = p0;
      plds[(r + 8 * half) * 32 + (lane & 15) + 16] = p1;
    }
    asm volatile("s_wait_dscnt 0x0" ::: "memory");
    // rebuild P (16x32) as a bf16 A-fragment from LDS
    union { v16bf v; unsigned short u[16]; } pa;
    const float* prow = plds + (lane & 15) * 32 + ((lane & 16) ? 8 : 0);
#pragma unroll
    for (int e = 0; e < 8; ++e) pa.u[e]     = f32_to_bf16(prow[e]);
#pragma unroll
    for (int e = 0; e < 8; ++e) pa.u[8 + e] = f32_to_bf16(prow[16 + e]);

    const unsigned short* vb =
        BvF + ((((size_t)h * 64 + j32) * 4) * 32 + lane) * 16;
    o0 = wmma_bf16(pa.v, *(const v16bf*)(vb       ), o0);
    o1 = wmma_bf16(pa.v, *(const v16bf*)(vb +  512), o1);
    o2 = wmma_bf16(pa.v, *(const v16bf*)(vb + 1024), o2);
    o3 = wmma_bf16(pa.v, *(const v16bf*)(vb + 1536), o3);
  }

  float* dst = attn + (size_t)c * SEQ * DIM;
  int n = lane & 15;
#pragma unroll
  for (int r = 0; r < 8; ++r) {
    int i = i_tile * 16 + r + 8 * half;
    float inv = 1.0f / lrow[r];
    float* row = dst + (size_t)i * DIM + h * 64;
    row[n]      = o0[r] * inv;
    row[n + 16] = o1[r] * inv;
    row[n + 32] = o2[r] * inv;
    row[n + 48] = o3[r] * inv;
  }
}

// ---------------- differential combine + RMSNorm + gamma -------------------
__global__ void combine_kernel(const float* __restrict__ attn,
                               const float* __restrict__ lq1,
                               const float* __restrict__ lk1,
                               const float* __restrict__ lq2,
                               const float* __restrict__ lk2,
                               const float* __restrict__ gamma,
                               float* __restrict__ wv) {
  int idx = blockIdx.x * blockDim.x + threadIdx.x;   // SEQ*NHEAD
  int h = idx & 15;
  int s = idx >> 4;
  float l1 = 0.f, l2 = 0.f;
#pragma unroll
  for (int i = 0; i < HDIM; ++i) { l1 += lq1[i] * lk1[i]; l2 += lq2[i] * lk2[i]; }
  const float LAMBDA_INIT = 0.8f - 0.6f * expf(-3.6f);
  float lam = expf(l1) - expf(l2) + LAMBDA_INIT;
  const float* a1 = attn + (size_t)s * DIM + h * 64;
  const float* a2 = attn + (size_t)SEQ * DIM + (size_t)s * DIM + h * 64;
  float w[64]; float acc = 0.f;
#pragma unroll
  for (int d = 0; d < 64; ++d) { float t = a1[d] - lam * a2[d]; w[d] = t; acc += t * t; }
  float rn = rsqrtf(acc * (1.0f / 64.0f) + 1e-5f) * (1.0f - LAMBDA_INIT);
  float* out = wv + (size_t)s * DIM + h * 64;
#pragma unroll
  for (int d = 0; d < 64; ++d) out[d] = w[d] * rn * gamma[d];
}

extern "C" void kernel_launch(void* const* d_in, const int* in_sizes, int n_in,
                              void* d_out, int out_size, void* d_ws, size_t ws_size,
                              hipStream_t stream) {
  const float* x   = (const float*)d_in[0];
  const float* Wq  = (const float*)d_in[1];
  const float* Wk  = (const float*)d_in[2];
  const float* Wv  = (const float*)d_in[3];
  const float* Wo  = (const float*)d_in[4];
  const float* lq1 = (const float*)d_in[5];
  const float* lk1 = (const float*)d_in[6];
  const float* lq2 = (const float*)d_in[7];
  const float* lk2 = (const float*)d_in[8];
  const float* gam = (const float*)d_in[9];

  char* ws = (char*)d_ws;
  size_t off = 0;
  auto alloc = [&](size_t bytes) { void* p = ws + off; off += (bytes + 255) & ~(size_t)255; return p; };

  unsigned short* xF   = (unsigned short*)alloc((size_t)SEQ * DIM * 2);
  unsigned short* wqF  = (unsigned short*)alloc((size_t)DIM * DIM * 2);
  unsigned short* wkF  = (unsigned short*)alloc((size_t)DIM * DIM * 2);
  unsigned short* wvF  = (unsigned short*)alloc((size_t)DIM * DIM * 2);
  unsigned short* woF  = (unsigned short*)alloc((size_t)DIM * DIM * 2);
  float*          Qf   = (float*)alloc((size_t)SEQ * DIM * 4);
  float*          Kf   = (float*)alloc((size_t)SEQ * DIM * 4);
  float*          Vf   = (float*)alloc((size_t)SEQ * DIM * 4);
  unsigned short* AqF  = (unsigned short*)alloc((size_t)SEQ * DIM * 2);
  unsigned short* BkF  = (unsigned short*)alloc((size_t)SEQ * DIM * 2);
  unsigned short* BvF  = (unsigned short*)alloc((size_t)SEQ * DIM * 2);
  float*          attn = (float*)alloc((size_t)2 * SEQ * DIM * 4);
  float*          wvb  = (float*)alloc((size_t)SEQ * DIM * 4);
  unsigned short* wvFr = (unsigned short*)alloc((size_t)SEQ * DIM * 2);

  // 1) pack activations and weights to WMMA fragments (bf16)
  pack_a_kernel<<<512, 256, 0, stream>>>(x, xF, DIM);
  pack_w_kernel<<<256, 256, 0, stream>>>(Wq, wqF);
  pack_w_kernel<<<256, 256, 0, stream>>>(Wk, wkF);
  pack_w_kernel<<<256, 256, 0, stream>>>(Wv, wvF);
  pack_w_kernel<<<256, 256, 0, stream>>>(Wo, woF);

  // 2) projections: Q/K/V = x @ W^T  (32x64 register-blocked WMMA GEMM)
  gemm_bf16_kernel<<<128, 256, 0, stream>>>(xF, wqF, Qf, 32, DIM);
  gemm_bf16_kernel<<<128, 256, 0, stream>>>(xF, wkF, Kf, 32, DIM);
  gemm_bf16_kernel<<<128, 256, 0, stream>>>(xF, wvF, Vf, 32, DIM);

  // 3) RoPE + repack Q->A-frags, K->B-frags; V->B-frags
  rope_pack_qk_kernel<<<512, 256, 0, stream>>>(Qf, Kf, AqF, BkF);
  pack_v_kernel<<<512, 256, 0, stream>>>(Vf, BvF);

  // 4) flash attention (both differential components), fp32 out
  flash_kernel<<<512, 256, 0, stream>>>(AqF, BkF, BvF, attn);

  // 5) differential combine + RMSNorm + gamma
  combine_kernel<<<128, 256, 0, stream>>>(attn, lq1, lk1, lq2, lk2, gam, wvb);

  // 6) output projection: out = wv @ Wo^T
  pack_a_kernel<<<512, 256, 0, stream>>>(wvb, wvFr, DIM);
  gemm_bf16_kernel<<<128, 256, 0, stream>>>(wvFr, woF, (float*)d_out, 32, DIM);
}
